// NN_58686433132630
// MI455X (gfx1250) — matrix operand order repaired and checked by reading, not compiled, not run on hardware
//
#include <hip/hip_runtime.h>
#include <hip/hip_bf16.h>

// ---------------------------------------------------------------------------
// B=512, D_IN=16384, D_LAT=4096.
// GEMM1: [512,16384]x[16384,4096]; GEMM2: [512,4096]x[4096,16384]
// ~600MB HBM traffic / 23.3 TB/s => memory-bound; bf16 WMMA (f32 accum)
// keeps the matrix path off the critical path. Double-buffered LDS hides
// global-load latency under the 8-WMMA compute burst with 1 barrier/K-step.
// ---------------------------------------------------------------------------

typedef __attribute__((ext_vector_type(16))) __bf16 v16bf;
typedef __attribute__((ext_vector_type(8)))  float  v8f;

#define BM 128
#define BN 128
#define BK 32
#define LDH 40              // padded LDS row stride in halfwords (80B, 16B-aligned)
#define NBINS 255
#define ALPHA 0.3f
#define EPS_BN 1e-3f

union FragAB { v16bf v; uint4 u[2]; };

__device__ __forceinline__ unsigned short f2bf(float f) {
    unsigned int u = __float_as_uint(f);
    u += 0x7FFFu + ((u >> 16) & 1u);            // round-to-nearest-even
    return (unsigned short)(u >> 16);
}
__device__ __forceinline__ unsigned int pack2bf(float x, float y) {
    unsigned int ux = __float_as_uint(x), uy = __float_as_uint(y);
    ux += 0x7FFFu + ((ux >> 16) & 1u);
    uy += 0x7FFFu + ((uy >> 16) & 1u);
    return (ux >> 16) | (uy & 0xFFFF0000u);
}

// ---------------------------------------------------------------------------
// GEMM: C[M,N] = A[M,K] * Bm[K,N], f32 in/out, bf16 WMMA inner product.
// 256 threads = 8 waves; 128x128 tile; each wave owns a 64x32 sub-tile.
// Double-buffered LDS: glbload(next) -> compute(cur) -> stage(next) -> barrier.
// ---------------------------------------------------------------------------
__global__ __launch_bounds__(256) void gemm_bf16_wmma(
    const float* __restrict__ A, const float* __restrict__ Bm,
    float* __restrict__ C, int M, int N, int K)
{
    __shared__ unsigned short sA[2][BM * LDH];   // [m][k] bf16
    __shared__ unsigned short sB[2][BN * LDH];   // [n][k] bf16 (transposed W tile)

    const int tid  = threadIdx.x;
    const int lane = tid & 31;
    const int wave = tid >> 5;
    const int wm   = wave & 1;     // 0..1 : 64-row slab
    const int wn   = wave >> 1;    // 0..3 : 32-col slab
    const int m0   = blockIdx.y * BM;
    const int n0   = blockIdx.x * BN;
    const int lh   = lane >> 4;    // lane half: 0 or 1
    const int l16  = lane & 15;

    // staging coordinates (hoisted; k0 added as offset to fixed base pointers)
    const int ar  = tid >> 3;            // A tile row   0..31 (+32 per pass)
    const int ac4 = (tid & 7) << 2;      // A tile col   0,4,...,28
    const int bk  = tid >> 5;            // W tile k-row 0..7  (+8 per pass)
    const int bc4 = (tid & 31) << 2;     // W tile col   0,4,...,124

    const float* Aptr = A  + (size_t)(m0 + ar) * K + ac4;   // k0 added in glbload
    const float* Bptr = Bm + (size_t)bk * N + n0 + bc4;     // k0 folded in below

    // fragment LDS base offsets (halfword units)
    int aOff[4], bOff[2];
#pragma unroll
    for (int i = 0; i < 4; ++i) aOff[i] = (wm * 64 + 16 * i + l16) * LDH + lh * 8;
#pragma unroll
    for (int j = 0; j < 2; ++j) bOff[j] = (wn * 32 + 16 * j + l16) * LDH + lh * 8;

    v8f acc[4][2];
#pragma unroll
    for (int i = 0; i < 4; ++i)
#pragma unroll
        for (int j = 0; j < 2; ++j)
#pragma unroll
            for (int v = 0; v < 8; ++v) acc[i][j][v] = 0.0f;

    float4 fa[4], fb[4];

    auto glbload = [&](int k0) {
#pragma unroll
        for (int p = 0; p < 4; ++p)
            fa[p] = *(const float4*)(Aptr + (size_t)(32 * p) * K + k0);
#pragma unroll
        for (int p = 0; p < 4; ++p)
            fb[p] = *(const float4*)(Bptr + (size_t)(k0 + 8 * p) * N);
    };

    auto stage = [&](int buf) {
#pragma unroll
        for (int p = 0; p < 4; ++p) {
            uint2 pk;
            pk.x = pack2bf(fa[p].x, fa[p].y);
            pk.y = pack2bf(fa[p].z, fa[p].w);
            *(uint2*)&sA[buf][(ar + 32 * p) * LDH + ac4] = pk;
        }
#pragma unroll
        for (int p = 0; p < 4; ++p) {
            const int kk = bk + 8 * p;
            sB[buf][(bc4 + 0) * LDH + kk] = f2bf(fb[p].x);
            sB[buf][(bc4 + 1) * LDH + kk] = f2bf(fb[p].y);
            sB[buf][(bc4 + 2) * LDH + kk] = f2bf(fb[p].z);
            sB[buf][(bc4 + 3) * LDH + kk] = f2bf(fb[p].w);
        }
    };

    auto compute = [&](int buf) {
        // A 16x32 layout: lanes 0-15 hold K 0-7 & 16-23; lanes 16-31 K 8-15 & 24-31
        FragAB a[4];
#pragma unroll
        for (int i = 0; i < 4; ++i) {
            const unsigned short* base = &sA[buf][aOff[i]];
            a[i].u[0] = *(const uint4*)(base);
            a[i].u[1] = *(const uint4*)(base + 16);
        }
        // B 32x16 layout: lanes 0-15 hold K 0-7 & 16-23; lanes 16-31 K 8-15 & 24-31
        FragAB b[2];
#pragma unroll
        for (int j = 0; j < 2; ++j) {
            const unsigned short* base = &sB[buf][bOff[j]];
            b[j].u[0] = *(const uint4*)(base);
            b[j].u[1] = *(const uint4*)(base + 16);
        }
#pragma unroll
        for (int i = 0; i < 4; ++i)
#pragma unroll
            for (int j = 0; j < 2; ++j)
                acc[i][j] = __builtin_amdgcn_wmma_f32_16x16x32_bf16(
                    false, a[i].v, false, b[j].v, (short)0, acc[i][j], false, false);
    };

    // prologue: fill buffer 0
    glbload(0);
    stage(0);
    __syncthreads();

    int cur = 0;
    for (int k0 = BK; k0 < K; k0 += BK) {
        glbload(k0);                       // next tile in flight
        if (k0 + BK < K) {                 // prefetch tile-after-next (global_prefetch_b8)
            __builtin_prefetch(Aptr + k0 + BK, 0, 1);
            __builtin_prefetch(Bptr + (size_t)(k0 + BK) * N, 0, 1);
        }
        compute(cur);                      // 8 WMMAs hide the load latency
        stage(cur ^ 1);                    // write other buffer (no reader conflict)
        __syncthreads();                   // single barrier per K-step
        cur ^= 1;
    }
    compute(cur);

    // epilogue: C/D layout: VGPR v, lane l -> M = v + (l/16)*8, N = l%16
#pragma unroll
    for (int i = 0; i < 4; ++i)
#pragma unroll
        for (int j = 0; j < 2; ++j) {
            const int col = n0 + wn * 32 + 16 * j + l16;
#pragma unroll
            for (int v = 0; v < 8; ++v) {
                const int row = m0 + wm * 64 + 16 * i + lh * 8 + v;
                C[(size_t)row * N + col] = acc[i][j][v];
            }
        }
}

// ---------------------------------------------------------------------------
// Column stats -> fold into per-column scale/shift (biased variance, BN train)
// ---------------------------------------------------------------------------
__global__ void colstats_kernel(const float* __restrict__ C, int M, int N,
                                const float* __restrict__ gamma,
                                const float* __restrict__ beta,
                                float* __restrict__ scale, float* __restrict__ shift)
{
    const int n = blockIdx.x * blockDim.x + threadIdx.x;
    if (n >= N) return;
    float s = 0.f, s2 = 0.f;
    for (int r = 0; r < M; ++r) {
        const float v = C[(size_t)r * N + n];
        s += v; s2 += v * v;
    }
    const float invM  = 1.0f / (float)M;
    const float mean  = s * invM;
    const float var   = fmaxf(s2 * invM - mean * mean, 0.0f);
    const float sc    = gamma[n] * rsqrtf(var + EPS_BN);
    scale[n] = sc;
    shift[n] = beta[n] - mean * sc;
}

// ---------------------------------------------------------------------------
// In-place BN + LeakyReLU, vectorized float4 (N % 4 == 0)
// ---------------------------------------------------------------------------
__global__ void bn_lrelu_kernel(float* __restrict__ data,
                                const float* __restrict__ scale,
                                const float* __restrict__ shift,
                                int N, long total4)
{
    long i = (long)blockIdx.x * blockDim.x + threadIdx.x;
    const long stride = (long)gridDim.x * blockDim.x;
    for (; i < total4; i += stride) {
        float4 h = ((const float4*)data)[i];
        const int n = (int)((i * 4) % N);
        float4 r;
        r.x = h.x * scale[n + 0] + shift[n + 0];
        r.y = h.y * scale[n + 1] + shift[n + 1];
        r.z = h.z * scale[n + 2] + shift[n + 2];
        r.w = h.w * scale[n + 3] + shift[n + 3];
        r.x = (r.x >= 0.f) ? r.x : ALPHA * r.x;
        r.y = (r.y >= 0.f) ? r.y : ALPHA * r.y;
        r.z = (r.z >= 0.f) ? r.z : ALPHA * r.z;
        r.w = (r.w >= 0.f) ? r.w : ALPHA * r.w;
        ((float4*)data)[i] = r;
    }
}

// ---------------------------------------------------------------------------
// EMD helpers: two-pass deterministic min/max over concat(xe, y)
// ---------------------------------------------------------------------------
__global__ void minmax_part_kernel(const float* __restrict__ a, long na,
                                   const float* __restrict__ b, long nb,
                                   float* __restrict__ part)
{
    __shared__ float smin[256], smax[256];
    const int tid = threadIdx.x;
    float lo = 3.402823466e38f, hi = -3.402823466e38f;
    const long total = na + nb;
    for (long i = (long)blockIdx.x * 256 + tid; i < total; i += (long)gridDim.x * 256) {
        const float v = (i < na) ? a[i] : b[i - na];
        lo = fminf(lo, v); hi = fmaxf(hi, v);
    }
    smin[tid] = lo; smax[tid] = hi;
    __syncthreads();
    for (int s = 128; s > 0; s >>= 1) {
        if (tid < s) {
            smin[tid] = fminf(smin[tid], smin[tid + s]);
            smax[tid] = fmaxf(smax[tid], smax[tid + s]);
        }
        __syncthreads();
    }
    if (tid == 0) { part[2 * blockIdx.x] = smin[0]; part[2 * blockIdx.x + 1] = smax[0]; }
}

__global__ void minmax_final_kernel(const float* __restrict__ part, int nparts,
                                    float* __restrict__ lohi)
{
    __shared__ float smin[256], smax[256];
    const int tid = threadIdx.x;
    float lo = 3.402823466e38f, hi = -3.402823466e38f;
    for (int i = tid; i < nparts; i += 256) {
        lo = fminf(lo, part[2 * i]); hi = fmaxf(hi, part[2 * i + 1]);
    }
    smin[tid] = lo; smax[tid] = hi;
    __syncthreads();
    for (int s = 128; s > 0; s >>= 1) {
        if (tid < s) {
            smin[tid] = fminf(smin[tid], smin[tid + s]);
            smax[tid] = fmaxf(smax[tid], smax[tid + s]);
        }
        __syncthreads();
    }
    if (tid == 0) { lohi[0] = smin[0]; lohi[1] = smax[0]; }
}

__global__ void hist_zero_kernel(unsigned int* __restrict__ h, int n)
{
    const int i = blockIdx.x * blockDim.x + threadIdx.x;
    if (i < n) h[i] = 0u;
}

__global__ void hist_kernel(const float* __restrict__ v, long n,
                            const float* __restrict__ lohi,
                            unsigned int* __restrict__ h)
{
    __shared__ unsigned int sh[NBINS];
    const int tid = threadIdx.x;
    for (int i = tid; i < NBINS; i += 256) sh[i] = 0u;
    __syncthreads();
    const float lo    = lohi[0];
    const float width = (lohi[1] - lo) / (float)NBINS;
    for (long i = (long)blockIdx.x * 256 + tid; i < n; i += (long)gridDim.x * 256) {
        int idx = (int)((v[i] - lo) / width);        // matches truncating int cast
        idx = min(max(idx, 0), NBINS - 1);
        atomicAdd(&sh[idx], 1u);
    }
    __syncthreads();
    for (int i = tid; i < NBINS; i += 256)
        if (sh[i]) atomicAdd(&h[i], sh[i]);
}

// ---------------------------------------------------------------------------
// Final EMD: one workgroup. Rank-sort 510 counts; searchsorted-by-counting.
// ---------------------------------------------------------------------------
__global__ void emd_final_kernel(const unsigned int* __restrict__ hx,
                                 const unsigned int* __restrict__ hy,
                                 float* __restrict__ loss_out)
{
    __shared__ float c[2 * NBINS];     // [0..254]=hx, [255..509]=hy (float counts)
    __shared__ float srt[2 * NBINS];   // sorted combined
    __shared__ float partial[256];
    const int tid = threadIdx.x;

    for (int i = tid; i < 2 * NBINS; i += 256)
        c[i] = (float)((i < NBINS) ? hx[i] : hy[i - NBINS]);
    __syncthreads();

    // stable rank sort of combined counts
    for (int i = tid; i < 2 * NBINS; i += 256) {
        const float v = c[i];
        int r = 0;
        for (int k = 0; k < 2 * NBINS; ++k)
            r += (c[k] < v) || (c[k] == v && k < i);
        srt[r] = v;
    }
    __syncthreads();

    // sum over j of ((ix-iy)/NBINS)^2 * (srt[j+1]-srt[j]);
    // ix = searchsorted(sort(hx), srt[j], 'right') = #{hx <= srt[j]}
    float acc = 0.f;
    for (int j = tid; j < 2 * NBINS - 1; j += 256) {
        const float v = srt[j];
        int ix = 0, iy = 0;
        for (int k = 0; k < NBINS; ++k) {
            ix += (c[k] <= v);
            iy += (c[NBINS + k] <= v);
        }
        const float d = (float)(ix - iy) * (1.0f / (float)NBINS);
        acc += d * d * (srt[j + 1] - srt[j]);
    }
    partial[tid] = acc;
    __syncthreads();
    for (int s = 128; s > 0; s >>= 1) {
        if (tid < s) partial[tid] += partial[tid + s];
        __syncthreads();
    }
    if (tid == 0) loss_out[0] = sqrtf(partial[0]);
}

// ---------------------------------------------------------------------------
extern "C" void kernel_launch(void* const* d_in, const int* in_sizes, int n_in,
                              void* d_out, int out_size, void* d_ws, size_t ws_size,
                              hipStream_t stream) {
    const int Bsz  = 512;
    const int DIN  = 16384;
    const int DLAT = 4096;

    const float* x       = (const float*)d_in[0];
    const float* y       = (const float*)d_in[1];
    const float* W_e     = (const float*)d_in[2];
    const float* gamma_e = (const float*)d_in[3];
    const float* beta_e  = (const float*)d_in[4];
    const float* W_g     = (const float*)d_in[5];
    const float* gamma_g = (const float*)d_in[6];
    const float* beta_g  = (const float*)d_in[7];

    float* gen  = (float*)d_out;                       // 512*16384 floats
    float* loss = gen + (size_t)Bsz * DIN;             // scalar at the tail

    // workspace layout
    float* w       = (float*)d_ws;
    float* xe      = w;                                // 512*4096
    float* scale1  = xe + (size_t)Bsz * DLAT;          // 4096
    float* shift1  = scale1 + DLAT;                    // 4096
    float* scale2  = shift1 + DLAT;                    // 16384
    float* shift2  = scale2 + DIN;                     // 16384
    float* part    = shift2 + DIN;                     // 512 (256 blocks x2)
    float* lohi    = part + 512;                       // 2
    unsigned int* hx = (unsigned int*)(lohi + 2);      // 255
    unsigned int* hy = hx + NBINS;                     // 255

    // ---- encoder: x @ W_e -> xe (raw), stats, BN+LeakyReLU in place -------
    gemm_bf16_wmma<<<dim3(DLAT / BN, Bsz / BM), 256, 0, stream>>>(x, W_e, xe, Bsz, DLAT, DIN);
    colstats_kernel<<<(DLAT + 255) / 256, 256, 0, stream>>>(xe, Bsz, DLAT, gamma_e, beta_e, scale1, shift1);
    bn_lrelu_kernel<<<2048, 256, 0, stream>>>(xe, scale1, shift1, DLAT, (long)Bsz * DLAT / 4);

    // ---- generator: xe @ W_g -> gen (raw in d_out), stats, BN+LReLU -------
    gemm_bf16_wmma<<<dim3(DIN / BN, Bsz / BM), 256, 0, stream>>>(xe, W_g, gen, Bsz, DIN, DLAT);
    colstats_kernel<<<(DIN + 255) / 256, 256, 0, stream>>>(gen, Bsz, DIN, gamma_g, beta_g, scale2, shift2);
    bn_lrelu_kernel<<<4096, 256, 0, stream>>>(gen, scale2, shift2, DIN, (long)Bsz * DIN / 4);

    // ---- EMD(xe, y) --------------------------------------------------------
    const long nxe = (long)Bsz * DLAT;
    minmax_part_kernel<<<256, 256, 0, stream>>>(xe, nxe, y, nxe, part);
    minmax_final_kernel<<<1, 256, 0, stream>>>(part, 256, lohi);
    hist_zero_kernel<<<2, 256, 0, stream>>>(hx, 2 * NBINS);
    hist_kernel<<<512, 256, 0, stream>>>(xe, nxe, lohi, hx);
    hist_kernel<<<512, 256, 0, stream>>>(y, nxe, lohi, hy);
    emd_final_kernel<<<1, 256, 0, stream>>>(hx, hy, loss);
}
